// T5MHSA_6408091206135
// MI455X (gfx1250) — compile-verified, matrix operand order checked by reading, not run on hardware
//
#include <hip/hip_runtime.h>
#include <hip/hip_bf16.h>

typedef __attribute__((ext_vector_type(16))) _Float16 v16h;
typedef __attribute__((ext_vector_type(8)))  float    v8f;

#define BINS 32
#define MAX_DIST 128
#define EMB 1024
#define DQ 64
#define DV 64
#define NH 16
#define NB 8
#define SEQ 1024

union V16H { v16h v; uint4 u[2]; };
union H8 { _Float16 h[8]; uint4 u; };

__device__ inline v8f vzero8() {
  v8f z = {0.f, 0.f, 0.f, 0.f, 0.f, 0.f, 0.f, 0.f};
  return z;
}

__device__ inline v8f wmma_f16(v16h a, v16h b, v8f c) {
  return __builtin_amdgcn_wmma_f32_16x16x32_f16(
      /*neg_a=*/false, a, /*neg_b=*/false, b,
      /*c_mod=*/(short)0, c, /*reuse_a=*/false, /*reuse_b=*/false);
}

// Load a 16x32 f16 tile (row-major source, row stride `stride` halves) into the
// ISA A/B fragment layout: lane l holds row m=l&15; lanes<16 take K={0..7,16..23},
// lanes>=16 take K={8..15,24..31}. Two contiguous 16B loads per lane.
__device__ inline v16h load_a16(const _Float16* __restrict__ src, int stride) {
  const int lane = threadIdx.x & 31;
  const int m  = lane & 15;
  const int kl = (lane >> 4) << 3;
  const _Float16* p = src + (size_t)m * stride + kl;
  V16H r;
  r.u[0] = *(const uint4*)(p);
  r.u[1] = *(const uint4*)(p + 16);
  return r.v;
}

__device__ inline float rmax16(float v) {
  v = fmaxf(v, __shfl_xor(v, 1, 32));
  v = fmaxf(v, __shfl_xor(v, 2, 32));
  v = fmaxf(v, __shfl_xor(v, 4, 32));
  v = fmaxf(v, __shfl_xor(v, 8, 32));
  return v;
}
__device__ inline float rsum16(float v) {
  v += __shfl_xor(v, 1, 32);
  v += __shfl_xor(v, 2, 32);
  v += __shfl_xor(v, 4, 32);
  v += __shfl_xor(v, 8, 32);
  return v;
}

// ---------------- one-time fp32 -> f16 conversion (8 elems / thread) --------
__global__ __launch_bounds__(256) void k_cvt_f16(
    const float* __restrict__ src, _Float16* __restrict__ dst, int n) {
  int i = (blockIdx.x * 256 + threadIdx.x) * 8;
  if (i >= n) return;
  float4 f0 = *(const float4*)(src + i);
  float4 f1 = *(const float4*)(src + i + 4);
  H8 o;
  o.h[0] = (_Float16)f0.x; o.h[1] = (_Float16)f0.y;
  o.h[2] = (_Float16)f0.z; o.h[3] = (_Float16)f0.w;
  o.h[4] = (_Float16)f1.x; o.h[5] = (_Float16)f1.y;
  o.h[6] = (_Float16)f1.z; o.h[7] = (_Float16)f1.w;
  *(uint4*)(dst + i) = o.u;
}

// ---------------- register-blocked 32x64 GEMM core (A.B^T, f16, f32 acc) ----
// Per K-step: 2 A-tile loads + 4 B-tile loads -> 8 WMMAs.
__device__ inline void gemm_32x64(const _Float16* __restrict__ a0,
                                  const _Float16* __restrict__ w0,
                                  int K, v8f acc[2][4]) {
  for (int k = 0; k < K; k += 32) {
    __builtin_prefetch(a0 + k + 64, 0, 3);
    v16h A0 = load_a16(a0 + k, K);
    v16h A1 = load_a16(a0 + 16 * K + k, K);
#pragma unroll
    for (int j = 0; j < 4; ++j) {
      v16h Bj = load_a16(w0 + (size_t)j * 16 * K + k, K);
      acc[0][j] = wmma_f16(A0, Bj, acc[0][j]);
      acc[1][j] = wmma_f16(A1, Bj, acc[1][j]);
    }
  }
}

// ---------------- QK projection: y = x @ qk_w^T + qk_b; scatter into q/k ----
__global__ __launch_bounds__(256) void k_proj_qk(
    const _Float16* __restrict__ x16, const _Float16* __restrict__ w16,
    const float* __restrict__ bvec,
    _Float16* __restrict__ q16, _Float16* __restrict__ k16) {
  const int lane = threadIdx.x & 31;
  const int wave = threadIdx.x >> 5;
  const int Nt4 = (2 * DQ * NH) / 64;                // 32
  const int total = (NB * SEQ / 32) * Nt4;           // 256*32
  int tile = blockIdx.x * 8 + wave;
  if (tile >= total) return;
  const int mt2 = tile / Nt4, nt4 = tile % Nt4;

  v8f acc[2][4] = {{vzero8(), vzero8(), vzero8(), vzero8()},
                   {vzero8(), vzero8(), vzero8(), vzero8()}};
  gemm_32x64(x16 + (size_t)mt2 * 32 * EMB, w16 + (size_t)nt4 * 64 * EMB, EMB, acc);

  const int hi = lane >> 4, col = lane & 15;
#pragma unroll
  for (int i = 0; i < 2; ++i)
#pragma unroll
    for (int j = 0; j < 4; ++j) {
      int n = nt4 * 64 + j * 16 + col;
      float bb = bvec[n];
      int h = n >> 7, c = n & 127;                   // head, channel within 2*DQ
#pragma unroll
      for (int r = 0; r < 8; ++r) {
        int row = mt2 * 32 + i * 16 + r + 8 * hi;    // flat token id
        int bidx = row >> 10, li = row & (SEQ - 1);
        float val = acc[i][j][r] + bb;
        size_t off = ((size_t)(bidx * NH + h) * SEQ + li) * DQ + (c & 63);
        if (c < DQ) q16[off] = (_Float16)val;
        else        k16[off] = (_Float16)val;
      }
    }
}

// ---------------- V projection: store transposed as (B,H,DV,L) --------------
__global__ __launch_bounds__(256) void k_proj_v(
    const _Float16* __restrict__ x16, const _Float16* __restrict__ w16,
    const float* __restrict__ bvec, _Float16* __restrict__ vT16) {
  const int lane = threadIdx.x & 31;
  const int wave = threadIdx.x >> 5;
  const int Nt4 = (DV * NH) / 64;                    // 16
  const int total = (NB * SEQ / 32) * Nt4;           // 256*16
  int tile = blockIdx.x * 8 + wave;
  if (tile >= total) return;
  const int mt2 = tile / Nt4, nt4 = tile % Nt4;

  v8f acc[2][4] = {{vzero8(), vzero8(), vzero8(), vzero8()},
                   {vzero8(), vzero8(), vzero8(), vzero8()}};
  gemm_32x64(x16 + (size_t)mt2 * 32 * EMB, w16 + (size_t)nt4 * 64 * EMB, EMB, acc);

  const int hi = lane >> 4, col = lane & 15;
#pragma unroll
  for (int i = 0; i < 2; ++i)
#pragma unroll
    for (int j = 0; j < 4; ++j) {
      int n = nt4 * 64 + j * 16 + col;
      float bb = bvec[n];
      int h = n >> 6, d = n & 63;
#pragma unroll
      for (int r = 0; r < 8; ++r) {
        int row = mt2 * 32 + i * 16 + r + 8 * hi;
        int bidx = row >> 10, li = row & (SEQ - 1);
        vT16[((size_t)(bidx * NH + h) * DV + d) * SEQ + li] =
            (_Float16)(acc[i][j][r] + bb);
      }
    }
}

// ---- async stage one 32-key chunk of K (32x64) and V^T (64x32) into LDS ----
// 256 threads x 16B cover each 4KB chunk; tracked by ASYNCcnt.
__device__ inline void stage_chunk(const _Float16* __restrict__ kg,
                                   const _Float16* __restrict__ vg,
                                   _Float16* sk, _Float16* sv, int tid) {
  // K chunk: 32 rows x 64 halves, contiguous 4KB
  unsigned koff = (unsigned)(uintptr_t)sk + (unsigned)tid * 16u;
  const _Float16* kga = kg + tid * 8;
  asm volatile("global_load_async_to_lds_b128 %0, %1, off"
               :: "v"(koff), "v"(kga) : "memory");
  // V chunk: 64 segments (d) x 32 halves (j), segment stride SEQ in memory
  const int seg = tid >> 2;
  const int wi = (tid & 3) * 8;
  unsigned voff = (unsigned)(uintptr_t)sv + (unsigned)(seg * 64 + wi * 2);
  const _Float16* vga = vg + (size_t)seg * SEQ + wi;
  asm volatile("global_load_async_to_lds_b128 %0, %1, off"
               :: "v"(voff), "v"(vga) : "memory");
}

// ---------------- Flash attention with T5 relative bias ---------------------
__global__ __launch_bounds__(256) void k_attn(
    const _Float16* __restrict__ q16, const _Float16* __restrict__ k16,
    const _Float16* __restrict__ vT16, const float* __restrict__ bias,
    _Float16* __restrict__ attn16) {
  __shared__ float s_fbias[MAX_DIST + 1];            // bias[bucket(d), h] folded
  __shared__ _Float16 s_p[8][16][40];                // per-wave P staging tile
  __shared__ _Float16 s_k[2][32][64];                // double-buffered K chunk
  __shared__ _Float16 s_v[2][64][32];                // double-buffered V^T chunk

  const int qblk = blockIdx.x;                       // 0..7 (128 rows each)
  const int bh = blockIdx.y;                         // 0..127
  const int b = bh >> 4, h = bh & 15;
  const int tid = threadIdx.x;
  const int lane = tid & 31;
  const int wave = tid >> 5;

  if (tid <= MAX_DIST) {
    int n = tid;
    int lo = (n < 16) ? n : 16;
    int nm = (n > 16) ? n : 16;
    int bucket = lo + (int)(logf((float)(nm - 15)) * (15.0f / logf(113.0f)));
    if (bucket > BINS - 1) bucket = BINS - 1;
    s_fbias[n] = bias[bucket * NH + h];
  }

  const int q0w = qblk * 128 + wave * 16;            // this wave's first q row
  const int hi = lane >> 4, col = lane & 15;
  const int klo = (lane >> 4) << 3;

  const _Float16* qb = q16 + ((size_t)(b * NH + h) * SEQ + q0w) * DQ;
  const _Float16* kbase = k16 + (size_t)(b * NH + h) * SEQ * DQ;
  const _Float16* vbase = vT16 + (size_t)(b * NH + h) * DV * SEQ;

  // stage chunk 0 into buffer 0 while we also load Q
  stage_chunk(kbase, vbase, &s_k[0][0][0], &s_v[0][0][0], tid);

  const v16h aq0 = load_a16(qb, DQ);                 // q cols 0..31
  const v16h aq1 = load_a16(qb + 32, DQ);            // q cols 32..63

  float m_i[8], l_i[8];
  v8f acc[4] = {vzero8(), vzero8(), vzero8(), vzero8()};
#pragma unroll
  for (int r = 0; r < 8; ++r) { m_i[r] = -1e30f; l_i[r] = 0.f; }

  asm volatile("s_wait_asynccnt 0x0" ::: "memory");
  __syncthreads();

  const float SCALE = 0.125f;                        // 1/sqrt(64)
  const int jt_end_w = (q0w + 15) >> 5;              // this wave's causal bound
  const int jt_blk = qblk * 4 + 3;                   // block's last chunk

  for (int jt = 0; jt <= jt_blk; ++jt) {
    const int cur = jt & 1;
    if (jt < jt_blk) {                               // prefetch next chunk
      const int j1 = (jt + 1) * 32;
      stage_chunk(kbase + (size_t)j1 * DQ, vbase + j1,
                  &s_k[cur ^ 1][0][0], &s_v[cur ^ 1][0][0], tid);
    }
    if (jt <= jt_end_w) {
      const int j0 = jt * 32;
      const _Float16* kt = &s_k[cur][0][0];
      // S(16x32) = Q(16x64) . K^T : 4 WMMAs (K fragments from LDS)
      v16h bk0a = load_a16(kt, 64);
      v16h bk0b = load_a16(kt + 32, 64);
      v16h bk1a = load_a16(kt + 16 * 64, 64);
      v16h bk1b = load_a16(kt + 16 * 64 + 32, 64);
      v8f s0 = vzero8();
      s0 = wmma_f16(aq0, bk0a, s0);
      s0 = wmma_f16(aq1, bk0b, s0);
      v8f s1 = vzero8();
      s1 = wmma_f16(aq0, bk1a, s1);
      s1 = wmma_f16(aq1, bk1b, s1);

      // online softmax with causal mask + folded T5 bias
#pragma unroll
      for (int r = 0; r < 8; ++r) {
        int row = q0w + r + 8 * hi;
        int c0 = j0 + col;
        int c1 = j0 + 16 + col;
        int sh0 = row - c0;
        int sh1 = row - c1;
        float v0 = (sh0 < 0) ? -1e30f
                   : s0[r] * SCALE + s_fbias[sh0 > MAX_DIST ? MAX_DIST : sh0];
        float v1 = (sh1 < 0) ? -1e30f
                   : s1[r] * SCALE + s_fbias[sh1 > MAX_DIST ? MAX_DIST : sh1];
        float mx = rmax16(fmaxf(v0, v1));
        float nm = fmaxf(m_i[r], mx);
        float corr = __expf(m_i[r] - nm);
        float p0 = __expf(v0 - nm);
        float p1 = __expf(v1 - nm);
        float rs = rsum16(p0 + p1);
        l_i[r] = l_i[r] * corr + rs;
        m_i[r] = nm;
        acc[0][r] *= corr; acc[1][r] *= corr; acc[2][r] *= corr; acc[3][r] *= corr;
        s_p[wave][r + 8 * hi][col] = (_Float16)p0;
        s_p[wave][r + 8 * hi][16 + col] = (_Float16)p1;
      }

      // Relayout P: C-fragment -> A-fragment via wave-private LDS tile
      const _Float16* prow = &s_p[wave][lane & 15][0];
      V16H ap;
      ap.u[0] = *(const uint4*)(prow + klo);
      ap.u[1] = *(const uint4*)(prow + klo + 16);
      v16h apv = ap.v;

      // O(16x64) += P(16x32) . V(32x64), V^T fragments from LDS: 4 WMMAs
#pragma unroll
      for (int t = 0; t < 4; ++t) {
        v16h bv = load_a16(&s_v[cur][t * 16][0], 32);
        acc[t] = wmma_f16(apv, bv, acc[t]);
      }
    }
    asm volatile("s_wait_asynccnt 0x0" ::: "memory");
    __syncthreads();
  }

  // epilogue: normalize, store f16 to (B, L, H*DV)
#pragma unroll
  for (int r = 0; r < 8; ++r) {
    float inv = 1.0f / l_i[r];
    int row = q0w + r + 8 * hi;
    _Float16* op = attn16 + ((size_t)(b * SEQ + row)) * (NH * DV) + h * DV + col;
    op[0]  = (_Float16)(acc[0][r] * inv);
    op[16] = (_Float16)(acc[1][r] * inv);
    op[32] = (_Float16)(acc[2][r] * inv);
    op[48] = (_Float16)(acc[3][r] * inv);
  }
}

// ---------------- Output projection: out = attn @ out_w^T + out_b ----------
__global__ __launch_bounds__(256) void k_proj_out(
    const _Float16* __restrict__ attn16, const _Float16* __restrict__ w16,
    const float* __restrict__ bvec, float* __restrict__ out) {
  const int lane = threadIdx.x & 31;
  const int wave = threadIdx.x >> 5;
  const int Nt4 = EMB / 64;                          // 16
  const int total = (NB * SEQ / 32) * Nt4;           // 256*16
  int tile = blockIdx.x * 8 + wave;
  if (tile >= total) return;
  const int mt2 = tile / Nt4, nt4 = tile % Nt4;

  v8f acc[2][4] = {{vzero8(), vzero8(), vzero8(), vzero8()},
                   {vzero8(), vzero8(), vzero8(), vzero8()}};
  gemm_32x64(attn16 + (size_t)mt2 * 32 * (NH * DV),
             w16 + (size_t)nt4 * 64 * (NH * DV), NH * DV, acc);

  const int hi = lane >> 4, col = lane & 15;
#pragma unroll
  for (int i = 0; i < 2; ++i)
#pragma unroll
    for (int j = 0; j < 4; ++j) {
      int n = nt4 * 64 + j * 16 + col;
      float bb = bvec[n];
#pragma unroll
      for (int r = 0; r < 8; ++r) {
        int row = mt2 * 32 + i * 16 + r + 8 * hi;
        out[(size_t)row * EMB + n] = acc[i][j][r] + bb;
      }
    }
}

extern "C" void kernel_launch(void* const* d_in, const int* in_sizes, int n_in,
                              void* d_out, int out_size, void* d_ws, size_t ws_size,
                              hipStream_t stream) {
  const float* x     = (const float*)d_in[0];
  const float* qk_w  = (const float*)d_in[1];
  const float* qk_b  = (const float*)d_in[2];
  const float* v_w   = (const float*)d_in[3];
  const float* v_b   = (const float*)d_in[4];
  const float* out_w = (const float*)d_in[5];
  const float* out_b = (const float*)d_in[6];
  const float* bias  = (const float*)d_in[7];
  float* out = (float*)d_out;

  const size_t PER = (size_t)NB * NH * SEQ * DQ;     // 8M halves = 16 MB
  _Float16* q16    = (_Float16*)d_ws;
  _Float16* k16    = q16 + PER;
  _Float16* vT16   = k16 + PER;
  _Float16* attn16 = vT16 + PER;
  _Float16* x16    = attn16 + PER;                   // 8M halves
  _Float16* qkw16  = x16 + (size_t)NB * SEQ * EMB;   // 2M halves
  _Float16* vw16   = qkw16 + (size_t)2 * DQ * NH * EMB;
  _Float16* outw16 = vw16 + (size_t)DV * NH * EMB;

  const int NX  = NB * SEQ * EMB;                    // 8M
  const int NQW = 2 * DQ * NH * EMB;                 // 2M
  const int NVW = DV * NH * EMB;                     // 1M
  // one-time conversions to f16
  k_cvt_f16<<<NX  / (256 * 8), 256, 0, stream>>>(x,     x16,    NX);
  k_cvt_f16<<<NQW / (256 * 8), 256, 0, stream>>>(qk_w,  qkw16,  NQW);
  k_cvt_f16<<<NVW / (256 * 8), 256, 0, stream>>>(v_w,   vw16,   NVW);
  k_cvt_f16<<<NVW / (256 * 8), 256, 0, stream>>>(out_w, outw16, NVW);

  // QK projection: 256 x 32 (32x64) tiles, 8 waves/block
  k_proj_qk<<<(256 * 32) / 8, 256, 0, stream>>>(x16, qkw16, qk_b, q16, k16);
  // V projection (stored transposed): 256 x 16 tiles
  k_proj_v<<<(256 * 16) / 8, 256, 0, stream>>>(x16, vw16, v_b, vT16);
  // Flash attention, async double-buffered K/V staging: (q-block, b*h)
  k_attn<<<dim3(SEQ / 128, NB * NH), 256, 0, stream>>>(q16, k16, vT16, bias, attn16);
  // Output projection
  k_proj_out<<<(256 * 16) / 8, 256, 0, stream>>>(attn16, outw16, out_b, out);
}
